// GPT2Attention_34445637714358
// MI455X (gfx1250) — matrix operand order repaired and checked by reading, not compiled, not run on hardware
//
#include <hip/hip_runtime.h>
#include <hip/hip_bf16.h>

// ---------------------------------------------------------------------------
// GPT-2 attention block for MI455X (gfx1250), wave32 + WMMA f16->f32 + TDM.
//   B=2, S=2048, H=1024, NH=16, HD=64
// ---------------------------------------------------------------------------

typedef __attribute__((ext_vector_type(16))) _Float16     v16h;
typedef __attribute__((ext_vector_type(8)))  float        v8f;
typedef __attribute__((ext_vector_type(4)))  unsigned int v4u;
typedef __attribute__((ext_vector_type(8)))  int          v8i;
typedef __attribute__((ext_vector_type(4)))  int          v4i;

union HV16 { v16h v; v4u u[2]; _Float16 h[16]; };

#define WMMA_F16(a, b, c) \
  __builtin_amdgcn_wmma_f32_16x16x32_f16(false, (a), false, (b), (short)0, (c), false, false)

#define Bsz 2
#define Ssz 2048
#define Hsz 1024
#define NH  16
#define HD  64

#if defined(__has_builtin)
#if __has_builtin(__builtin_amdgcn_tensor_load_to_lds) && \
    __has_builtin(__builtin_amdgcn_s_wait_tensorcnt)
#define HAVE_TDM 1
#endif
#endif

// ---------------------------------------------------------------------------
// Stage 1: f32 -> f16 conversion
// ---------------------------------------------------------------------------
__global__ void f32_to_f16_kernel(const float* __restrict__ in,
                                  _Float16* __restrict__ out, int n) {
  int i = blockIdx.x * blockDim.x + threadIdx.x;
  if (i < n) out[i] = (_Float16)in[i];
}

// ---------------------------------------------------------------------------
// Stage 2/4: tiled WMMA GEMM  C[M,N] = A[M,K] @ B[K,N] + bias
//   mode 0: write f32 to out (output projection)
//   mode 1: QKV epilogue -> scatter Q (scaled by 1/8), K head-major, V transposed
// Block: 256 threads = 8 waves (4x2), tile 128x128, K-step 32.
// A tile moved by the Tensor Data Mover (async DMA, TENSORcnt) when available.
// ---------------------------------------------------------------------------
#define BM 128
#define BN 128
#define BK 32
#define LDA 40   // BK + 8 pad: 80B row stride, 16B aligned (TDM pad reproduces this)
#define LDB 40

#ifdef HAVE_TDM
// Issue one TDM DMA: 128x32 f16 tile (rows of A) -> padded LDS [128][40].
// D# per CDNA5 ISA 8.3/8.4: group0 {count,lds_addr,global_addr,type=2},
// group1 {data_size=2B, pad every 16 DWORDs by 4 DWORDs, dims, strides}.
__device__ __forceinline__ void tdm_load_tile_a(const _Float16* gsrc,
                                                unsigned lds_off, int K) {
  v4u g0;
  g0[0] = 1u;                                   // count=1 (valid user descriptor)
  g0[1] = lds_off;                              // lds_addr (bytes)
  unsigned long long ga = (unsigned long long)(uintptr_t)gsrc;
  g0[2] = (unsigned)ga;                         // global_addr[31:0]
  g0[3] = (unsigned)(ga >> 32) | 0x80000000u;   // global_addr[56:32] | type=2<<30
  v8i g1;
  g1[0] = (1 << 16)      // data_size = 1 -> 2 bytes
        | (1 << 20)      // pad_enable
        | (3 << 22)      // pad_interval = 3 -> every 16 DWORDs (32 f16)
        | (3 << 25);     // pad_amount   = 3 -> 4 DWORDs (8 f16) => row stride 40
  g1[1] = (K & 0xFFFF) << 16;                          // tensor_dim0 lo16
  g1[2] = ((unsigned)K >> 16) | (0xFFFFu << 16);       // tensor_dim0 hi | dim1 lo (large)
  g1[3] = 0x7FFF | (BK << 16);                         // dim1 hi | tile_dim0 = 32
  g1[4] = BM;                                          // tile_dim1 = 128, tile_dim2 = 0
  g1[5] = K;                                           // tensor_dim0_stride lo32
  g1[6] = 0;                                           // stride hi | dim1_stride lo
  g1[7] = 0;
  v4i zz = {};
#if __clang_major__ >= 23
  v8i z8 = {};
  __builtin_amdgcn_tensor_load_to_lds(g0, g1, zz, zz, z8, 0);
#else
  __builtin_amdgcn_tensor_load_to_lds(g0, g1, zz, zz, 0);
#endif
}
#endif

__global__ __launch_bounds__(256) void gemm_wmma_f16(
    const _Float16* __restrict__ A, const _Float16* __restrict__ Bm,
    const float* __restrict__ bias, int M, int N, int K, int mode,
    float* __restrict__ out,
    _Float16* __restrict__ Qb, _Float16* __restrict__ Kb,
    _Float16* __restrict__ Vt) {
  __shared__ alignas(16) _Float16 As[BM][LDA];
  __shared__ alignas(16) _Float16 Bst[BN][LDB];   // B tile stored transposed [n][k]

  const int tid   = threadIdx.x;
  const int lane  = tid & 31;
  const int wave  = tid >> 5;
  const int wm    = wave & 3;   // 4 row-slabs of 32
  const int wn    = wave >> 2;  // 2 col-slabs of 64
  const int nlo   = lane & 15;
  const int khalf = lane >> 4;
  const size_t bm0 = (size_t)blockIdx.x * BM;
  const size_t bn0 = (size_t)blockIdx.y * BN;

  v8f zero = {};
  v8f acc[2][4];
#pragma unroll
  for (int mt = 0; mt < 2; ++mt)
#pragma unroll
    for (int nt = 0; nt < 4; ++nt) acc[mt][nt] = zero;

  for (int k0 = 0; k0 < K; k0 += BK) {
#ifdef HAVE_TDM
    // --- A tile via Tensor Data Mover: one DMA, issued by wave 0 only ---
    if (wave == 0) {
      tdm_load_tile_a(A + bm0 * (size_t)K + k0,
                      (unsigned)(unsigned long long)(uintptr_t)&As[0][0], K);
      __builtin_amdgcn_s_wait_tensorcnt(0);
    }
#else
    // --- A tile: 128x32, each thread copies 16 contiguous f16 (two 16B) ---
    {
      const int r = tid >> 1;
      const int c = (tid & 1) * 16;
      const _Float16* src = A + (bm0 + r) * (size_t)K + k0 + c;
      *(v4u*)&As[r][c]     = *(const v4u*)(src);
      *(v4u*)&As[r][c + 8] = *(const v4u*)(src + 8);
    }
#endif
    // --- B tile: 32x128, read coalesced, store transposed ---
    {
      const int r = tid >> 3;
      const int c = (tid & 7) * 16;
      const _Float16* src = Bm + (size_t)(k0 + r) * N + bn0 + c;
      HV16 t;
      t.u[0] = *(const v4u*)(src);
      t.u[1] = *(const v4u*)(src + 8);
#pragma unroll
      for (int j = 0; j < 16; ++j) Bst[c + j][r] = t.h[j];
    }
    __syncthreads();

    // --- fragments in contiguous k order (k = 16*khalf + s), same for A and B ---
    HV16 af[2], bf[4];
#pragma unroll
    for (int mt = 0; mt < 2; ++mt) {
      const _Float16* p = &As[wm * 32 + mt * 16 + nlo][0];
      af[mt].u[0] = *(const v4u*)(p + khalf * 16);       // K = khalf*16 .. +7
      af[mt].u[1] = *(const v4u*)(p + khalf * 16 + 8);   // K = khalf*16+8 .. +15
    }
#pragma unroll
    for (int nt = 0; nt < 4; ++nt) {
      const _Float16* p = &Bst[wn * 64 + nt * 16 + nlo][0];
      bf[nt].u[0] = *(const v4u*)(p + khalf * 16);       // K = khalf*16 .. +7
      bf[nt].u[1] = *(const v4u*)(p + khalf * 16 + 8);   // K = khalf*16+8 .. +15
    }
#pragma unroll
    for (int mt = 0; mt < 2; ++mt)
#pragma unroll
      for (int nt = 0; nt < 4; ++nt)
        acc[mt][nt] = WMMA_F16(af[mt].v, bf[nt].v, acc[mt][nt]);
    __syncthreads();
  }

  // --- epilogue ---
  const int rbase = khalf * 8;
  if (mode == 0) {
#pragma unroll
    for (int mt = 0; mt < 2; ++mt)
#pragma unroll
      for (int nt = 0; nt < 4; ++nt) {
        const int col = (int)bn0 + wn * 64 + nt * 16 + nlo;
        const float bv = bias[col];
#pragma unroll
        for (int i = 0; i < 8; ++i) {
          const size_t row = bm0 + wm * 32 + mt * 16 + rbase + i;
          out[row * (size_t)N + col] = acc[mt][nt][i] + bv;
        }
      }
  } else {
    // QKV scatter; part is block-uniform because BN=128 divides 1024.
    const int part = (int)(bn0 >> 10);  // 0=Q 1=K 2=V
#pragma unroll
    for (int mt = 0; mt < 2; ++mt)
#pragma unroll
      for (int nt = 0; nt < 4; ++nt) {
        const int col = (int)bn0 + wn * 64 + nt * 16 + nlo;
        const float bv = bias[col];
        const int w  = col & 1023;
        const int hh = w >> 6;
        const int d  = w & 63;
#pragma unroll
        for (int i = 0; i < 8; ++i) {
          const size_t row = bm0 + wm * 32 + mt * 16 + rbase + i;
          const float v = acc[mt][nt][i] + bv;
          const size_t bh = (size_t)(row >> 11) * NH + hh;
          const int s = (int)(row & 2047);
          if (part == 0)
            Qb[(bh * Ssz + s) * HD + d] = (_Float16)(v * 0.125f);  // fold 1/sqrt(64)
          else if (part == 1)
            Kb[(bh * Ssz + s) * HD + d] = (_Float16)v;
          else
            Vt[(bh * HD + d) * Ssz + s] = (_Float16)v;             // V transposed
        }
      }
  }
}

// ---------------------------------------------------------------------------
// Stage 3: one-wave flash attention, 16 queries per block, 32-wide KV chunks.
// Q [BH][S][64] (pre-scaled), K [BH][S][64], V [BH][64][S], Ctx [B][S][H] f16.
// ---------------------------------------------------------------------------
__global__ __launch_bounds__(32) void attn_wmma(
    const _Float16* __restrict__ Q, const _Float16* __restrict__ Kk,
    const _Float16* __restrict__ Vt, _Float16* __restrict__ Ctx) {
  __shared__ alignas(16) _Float16 P[16][40];   // probs bounce buffer [m][k]

  const int lane  = threadIdx.x & 31;
  const int nlo   = lane & 15;
  const int khalf = lane >> 4;
  const int rbase = khalf * 8;
  const int qt = blockIdx.x;
  const int hh = blockIdx.y;
  const int bb = blockIdx.z;
  const size_t bh = (size_t)bb * NH + hh;
  const int q0 = qt * 16;

  const _Float16* Qp = Q  + bh * Ssz * HD;
  const _Float16* Kp = Kk + bh * Ssz * HD;
  const _Float16* Vp = Vt + bh * HD * Ssz;

  // Q fragments held in registers for the whole KV loop
  HV16 qa[2];
  {
    const _Float16* qrow = Qp + (size_t)(q0 + nlo) * HD;
#pragma unroll
    for (int kk = 0; kk < 2; ++kk) {
      qa[kk].u[0] = *(const v4u*)(qrow + kk * 32 + khalf * 16);
      qa[kk].u[1] = *(const v4u*)(qrow + kk * 32 + khalf * 16 + 8);
    }
  }

  v8f zero = {};
  v8f acc[4];
#pragma unroll
  for (int nt = 0; nt < 4; ++nt) acc[nt] = zero;
  float m_i[8], l_i[8];
#pragma unroll
  for (int i = 0; i < 8; ++i) { m_i[i] = -1e30f; l_i[i] = 0.0f; }

  const int nchunks = (q0 + 15) / 32 + 1;   // causal: only chunks touching k<=q
  for (int c = 0; c < nchunks; ++c) {
    const int kv0 = c * 32;

    // Prefetch next chunk's K/V lines while this chunk computes.
    if (c + 1 < nchunks) {
      const _Float16* nk = Kp + (size_t)(kv0 + 32 + nlo) * HD;
      __builtin_prefetch(nk, 0, 0);
      __builtin_prefetch(nk + 16 * HD, 0, 0);
      __builtin_prefetch(Vp + (size_t)nlo * Ssz + kv0 + 32, 0, 0);
    }

    // --- scores: S = Q @ K^T (K rows are B-fragments: contiguous loads) ---
    v8f s0 = zero, s1 = zero;
    {
      const _Float16* k0row = Kp + (size_t)(kv0 + nlo) * HD;
      const _Float16* k1row = Kp + (size_t)(kv0 + 16 + nlo) * HD;
      HV16 kb;
#pragma unroll
      for (int kk = 0; kk < 2; ++kk) {
        kb.u[0] = *(const v4u*)(k0row + kk * 32 + khalf * 16);
        kb.u[1] = *(const v4u*)(k0row + kk * 32 + khalf * 16 + 8);
        s0 = WMMA_F16(qa[kk].v, kb.v, s0);
      }
#pragma unroll
      for (int kk = 0; kk < 2; ++kk) {
        kb.u[0] = *(const v4u*)(k1row + kk * 32 + khalf * 16);
        kb.u[1] = *(const v4u*)(k1row + kk * 32 + khalf * 16 + 8);
        s1 = WMMA_F16(qa[kk].v, kb.v, s1);
      }
    }

    // --- analytic causal mask + online softmax (row = rbase+i, col = lane) ---
    const int col0 = kv0 + nlo;
    const int col1 = kv0 + 16 + nlo;
#pragma unroll
    for (int i = 0; i < 8; ++i) {
      const int q = q0 + rbase + i;
      if (col0 > q) s0[i] = -1e30f;
      if (col1 > q) s1[i] = -1e30f;
    }
#pragma unroll
    for (int i = 0; i < 8; ++i) {
      float mx = fmaxf(s0[i], s1[i]);
      mx = fmaxf(mx, __shfl_xor(mx, 1, 32));
      mx = fmaxf(mx, __shfl_xor(mx, 2, 32));
      mx = fmaxf(mx, __shfl_xor(mx, 4, 32));
      mx = fmaxf(mx, __shfl_xor(mx, 8, 32));
      const float mnew = fmaxf(m_i[i], mx);
      const float corr = __expf(m_i[i] - mnew);
      m_i[i] = mnew;
      const float p0 = __expf(s0[i] - mnew);
      const float p1 = __expf(s1[i] - mnew);
      float rs = p0 + p1;
      rs += __shfl_xor(rs, 1, 32);
      rs += __shfl_xor(rs, 2, 32);
      rs += __shfl_xor(rs, 4, 32);
      rs += __shfl_xor(rs, 8, 32);
      l_i[i] = l_i[i] * corr + rs;
#pragma unroll
      for (int nt = 0; nt < 4; ++nt) acc[nt][i] *= corr;
      P[rbase + i][nlo]      = (_Float16)p0;
      P[rbase + i][16 + nlo] = (_Float16)p1;
    }
    __syncthreads();

    // --- P re-layout to A-fragment, then ctx += P @ V ---
    HV16 pa;
    {
      const _Float16* pr = &P[nlo][0];
      pa.u[0] = *(const v4u*)(pr + khalf * 16);
      pa.u[1] = *(const v4u*)(pr + khalf * 16 + 8);
    }
#pragma unroll
    for (int nt = 0; nt < 4; ++nt) {
      HV16 vb;
      const _Float16* vr = Vp + (size_t)(nt * 16 + nlo) * Ssz + kv0;
      vb.u[0] = *(const v4u*)(vr + khalf * 16);
      vb.u[1] = *(const v4u*)(vr + khalf * 16 + 8);
      acc[nt] = WMMA_F16(pa.v, vb.v, acc[nt]);
    }
    __syncthreads();
  }

  // --- normalize and write merged-head context [B][S][H] as f16 ---
#pragma unroll
  for (int i = 0; i < 8; ++i) {
    const float inv = 1.0f / l_i[i];
    const int s = q0 + rbase + i;
    _Float16* dst = Ctx + ((size_t)bb * Ssz + s) * Hsz + hh * HD;
#pragma unroll
    for (int nt = 0; nt < 4; ++nt)
      dst[nt * 16 + nlo] = (_Float16)(acc[nt][i] * inv);
  }
}

// ---------------------------------------------------------------------------
// Host launcher
// ---------------------------------------------------------------------------
extern "C" void kernel_launch(void* const* d_in, const int* in_sizes, int n_in,
                              void* d_out, int out_size, void* d_ws, size_t ws_size,
                              hipStream_t stream) {
  const float* hidden = (const float*)d_in[0];
  // d_in[1] = attention_mask: causal by construction, applied analytically
  const float* W_attn = (const float*)d_in[2];
  const float* b_attn = (const float*)d_in[3];
  const float* W_proj = (const float*)d_in[4];
  const float* b_proj = (const float*)d_in[5];
  float* out = (float*)d_out;

  const size_t M   = (size_t)Bsz * Ssz;          // 4096
  const size_t nX  = M * Hsz;                    // 4,194,304
  const size_t nWa = (size_t)Hsz * 3 * Hsz;      // 3,145,728
  const size_t nWp = (size_t)Hsz * Hsz;          // 1,048,576
  const size_t nQ  = (size_t)Bsz * NH * Ssz * HD;

  char* ws = (char*)d_ws;
  size_t off = 0;
  auto take = [&](size_t bytes) { char* p = ws + off; off += (bytes + 255) & ~(size_t)255; return p; };
  _Float16* Xh  = (_Float16*)take(nX * 2);   // hidden f16; reused as Ctx after stage 2
  _Float16* Wah = (_Float16*)take(nWa * 2);
  _Float16* Wph = (_Float16*)take(nWp * 2);
  _Float16* Qb  = (_Float16*)take(nQ * 2);
  _Float16* Kb  = (_Float16*)take(nQ * 2);
  _Float16* Vt  = (_Float16*)take(nQ * 2);
  _Float16* Ctx = Xh;  // lifetime of Xh ends after QKV GEMM

  // Stage 1: conversions
  f32_to_f16_kernel<<<dim3((unsigned)((nX  + 255) / 256)), dim3(256), 0, stream>>>(hidden, Xh,  (int)nX);
  f32_to_f16_kernel<<<dim3((unsigned)((nWa + 255) / 256)), dim3(256), 0, stream>>>(W_attn, Wah, (int)nWa);
  f32_to_f16_kernel<<<dim3((unsigned)((nWp + 255) / 256)), dim3(256), 0, stream>>>(W_proj, Wph, (int)nWp);

  // Stage 2: fused QKV projection + head scatter (Q pre-scaled, V transposed)
  gemm_wmma_f16<<<dim3((unsigned)(M / BM), (3 * Hsz) / BN), dim3(256), 0, stream>>>(
      Xh, Wah, b_attn, (int)M, 3 * Hsz, Hsz, /*mode=*/1,
      nullptr, Qb, Kb, Vt);

  // Stage 3: flash attention, one wave per 16-query tile per (b, head)
  attn_wmma<<<dim3(Ssz / 16, NH, Bsz), dim3(32), 0, stream>>>(Qb, Kb, Vt, Ctx);

  // Stage 4: output projection -> f32
  gemm_wmma_f16<<<dim3((unsigned)(M / BM), Hsz / BN), dim3(256), 0, stream>>>(
      Ctx, Wph, b_proj, (int)M, Hsz, Hsz, /*mode=*/0,
      out, nullptr, nullptr, nullptr);
}
